// GraphClassifier_88648124991032
// MI455X (gfx1250) — matrix operand order repaired and checked by reading, not compiled, not run on hardware
//
#include <hip/hip_runtime.h>
#include <hip/hip_bf16.h>

#define NN   50000
#define NE   800000
#define EMBD 64
#define HIDD 64
#define NG   512
#define NC   10

typedef __attribute__((ext_vector_type(16))) __bf16 v16bf;
typedef __attribute__((ext_vector_type(8)))  float  v8f;

__device__ __forceinline__ unsigned short f2bf(float f) {
  unsigned int u = __float_as_uint(f);
  u += 0x7FFFu + ((u >> 16) & 1u);          // round-to-nearest-even
  return (unsigned short)(u >> 16);
}

__device__ __forceinline__ v16bf load_frag(const unsigned short* lo,
                                           const unsigned short* hi) {
  union { uint4 u[2]; v16bf v; } c;
  c.u[0] = *reinterpret_cast<const uint4*>(lo);
  c.u[1] = *reinterpret_cast<const uint4*>(hi);
  return c.v;
}

// ---------------- utility kernels ----------------

__global__ void zero_f32(float* p, int n) {
  int t = blockIdx.x * blockDim.x + threadIdx.x;
  if (t < n) p[t] = 0.0f;
}

__global__ void gather_embed(const int* __restrict__ ids,
                             const float* __restrict__ emb,
                             float* __restrict__ x) {
  int t = blockIdx.x * blockDim.x + threadIdx.x;
  if (t >= NN * EMBD) return;
  int i = t >> 6, c = t & 63;
  x[t] = emb[ids[i] * EMBD + c];
}

// convert 64x64 f32 weights -> transposed bf16 (WT[n][k] = W[k][n])
__global__ void prep_weights(const float* __restrict__ Wl0, const float* __restrict__ Wr0,
                             const float* __restrict__ Wl1, const float* __restrict__ Wr1,
                             unsigned short* __restrict__ WlT0, unsigned short* __restrict__ WrT0,
                             unsigned short* __restrict__ WlT1, unsigned short* __restrict__ WrT1) {
  int t = blockIdx.x * blockDim.x + threadIdx.x;
  if (t >= HIDD * HIDD) return;
  int k = t >> 6, n = t & 63;
  int o = n * HIDD + k;
  WlT0[o] = f2bf(Wl0[t]);
  WrT0[o] = f2bf(Wr0[t]);
  WlT1[o] = f2bf(Wl1[t]);
  WrT1[o] = f2bf(Wr1[t]);
}

// per (edge, 4-float chunk): s[dst] += x[src]; cnt[dst] += 1
__global__ void scatter_edges(const int* __restrict__ src, const int* __restrict__ dst,
                              const float* __restrict__ x,
                              float* __restrict__ s, float* __restrict__ cnt) {
  int t = blockIdx.x * blockDim.x + threadIdx.x;
  if (t >= NE * 16) return;
  int e = t >> 4, q = t & 15;
  int si = src[e], di = dst[e];
  const float4 v = *reinterpret_cast<const float4*>(x + si * HIDD + q * 4);
  float* p = s + di * HIDD + q * 4;
  atomicAdd(p + 0, v.x);
  atomicAdd(p + 1, v.y);
  atomicAdd(p + 2, v.z);
  atomicAdd(p + 3, v.w);
  if (q == 0) atomicAdd(cnt + di, 1.0f);
}

// mean = cnt>0 ? s/cnt : 0 -> bf16 ; x -> bf16
__global__ void mean_bf_kernel(const float* __restrict__ s, const float* __restrict__ cnt,
                               const float* __restrict__ x,
                               unsigned short* __restrict__ meanb,
                               unsigned short* __restrict__ xb) {
  int t = blockIdx.x * blockDim.x + threadIdx.x;
  if (t >= NN * HIDD) return;
  int i = t >> 6;
  float c = cnt[i];
  float m = (c > 0.0f) ? (s[t] / c) : 0.0f;
  meanb[t] = f2bf(m);
  xb[t]    = f2bf(x[t]);
}

// ---------------- WMMA fused SAGE GEMM ----------------
// out = relu(mean @ Wl + x @ Wr + b), HID=64.
// Weights (16 B-fragments, 128 VGPRs) hoisted once per wave; wave grid-strides
// over 16-row tiles so the 16 WMMAs per tile run back-to-back from registers.
__global__ __launch_bounds__(128, 1)
void gemm_sage(const unsigned short* __restrict__ meanb,
               const unsigned short* __restrict__ xb,
               const unsigned short* __restrict__ WlT,   // [64][64] transposed bf16
               const unsigned short* __restrict__ WrT,
               const float* __restrict__ bias,
               float* __restrict__ out, int nwaves) {
  const int lane   = threadIdx.x & 31;
  const int waveId = blockIdx.x * (blockDim.x >> 5) + (threadIdx.x >> 5);
  const int m      = lane & 15;
  const int half   = lane >> 4;                  // 0 | 1
  const int kbaseA = half * 8;                   // A frag K offset (ISA 16-bit A 16x32)
  const int kbaseB = half * 8;                   // B frag K offset (same order as A)

  // Hoist all weight fragments: [mat(l,r)][ct][ks]
  v16bf bw[2][4][2];
#pragma unroll
  for (int ct = 0; ct < 4; ++ct) {
    const int col = ct * 16 + m;
#pragma unroll
    for (int ks = 0; ks < 2; ++ks) {
      const unsigned short* pl = WlT + col * HIDD + ks * 32 + kbaseB;
      const unsigned short* pr = WrT + col * HIDD + ks * 32 + kbaseB;
      bw[0][ct][ks] = load_frag(pl, pl + 16);
      bw[1][ct][ks] = load_frag(pr, pr + 16);
    }
  }
  float bv[4];
#pragma unroll
  for (int ct = 0; ct < 4; ++ct) bv[ct] = bias[ct * 16 + m];

  const int ntiles = NN >> 4;                    // 3125
  for (int tile = waveId; tile < ntiles; tile += nwaves) {   // wave-uniform loop
    const int row0 = tile << 4;
    const int row  = row0 + m;

    v16bf amean[2], ax[2];
#pragma unroll
    for (int ks = 0; ks < 2; ++ks) {
      const unsigned short* pm = meanb + row * HIDD + ks * 32 + kbaseA;
      const unsigned short* px = xb    + row * HIDD + ks * 32 + kbaseA;
      amean[ks] = load_frag(pm, pm + 16);
      ax[ks]    = load_frag(px, px + 16);
    }

    v8f acc[4];
#pragma unroll
    for (int ct = 0; ct < 4; ++ct)
      acc[ct] = (v8f){bv[ct], bv[ct], bv[ct], bv[ct], bv[ct], bv[ct], bv[ct], bv[ct]};

#pragma unroll
    for (int ct = 0; ct < 4; ++ct) {
#pragma unroll
      for (int ks = 0; ks < 2; ++ks) {
        acc[ct] = __builtin_amdgcn_wmma_f32_16x16x32_bf16(
            false, amean[ks], false, bw[0][ct][ks], (short)0, acc[ct], false, false);
        acc[ct] = __builtin_amdgcn_wmma_f32_16x16x32_bf16(
            false, ax[ks], false, bw[1][ct][ks], (short)0, acc[ct], false, false);
      }
    }

    // store with ReLU; C/D layout: lane<16 -> M=r, lane>=16 -> M=r+8; N = lane&15
#pragma unroll
    for (int ct = 0; ct < 4; ++ct) {
      const int col = ct * 16 + m;
#pragma unroll
      for (int r = 0; r < 8; ++r) {
        const int mrow = half ? (r + 8) : r;
        float v = acc[ct][r];
        out[(row0 + mrow) * HIDD + col] = (v > 0.0f) ? v : 0.0f;
      }
    }
  }
}

// ---------------- pooling + head ----------------

__global__ void pool_scatter(const int* __restrict__ batch, const float* __restrict__ x,
                             float* __restrict__ pooled, float* __restrict__ gcnt) {
  int t = blockIdx.x * blockDim.x + threadIdx.x;
  if (t >= NN * 16) return;
  int i = t >> 4, q = t & 15;
  int g = batch[i];
  const float4 v = *reinterpret_cast<const float4*>(x + i * HIDD + q * 4);
  float* p = pooled + g * HIDD + q * 4;
  atomicAdd(p + 0, v.x);
  atomicAdd(p + 1, v.y);
  atomicAdd(p + 2, v.z);
  atomicAdd(p + 3, v.w);
  if (q == 0) atomicAdd(gcnt + g, 1.0f);
}

__global__ void final_out(const float* __restrict__ pooled, const float* __restrict__ gcnt,
                          const float* __restrict__ Wout, const float* __restrict__ bout,
                          float* __restrict__ out) {
  int g = blockIdx.x;
  int j = threadIdx.x;
  if (j >= NC) return;
  float c = gcnt[g];
  float inv = (c > 0.0f) ? (1.0f / c) : 0.0f;
  float sum = bout[j];
  for (int k = 0; k < HIDD; ++k)
    sum += pooled[g * HIDD + k] * inv * Wout[k * NC + j];
  out[g * NC + j] = sum;
}

// ---------------- host launcher ----------------

extern "C" void kernel_launch(void* const* d_in, const int* in_sizes, int n_in,
                              void* d_out, int out_size, void* d_ws, size_t ws_size,
                              hipStream_t stream) {
  (void)in_sizes; (void)n_in; (void)out_size; (void)ws_size;

  const int*   node_ids = (const int*)d_in[0];
  const int*   edge     = (const int*)d_in[1];     // [2, NE]
  const int*   batch    = (const int*)d_in[2];
  const float* emb      = (const float*)d_in[3];
  const float* Wl0      = (const float*)d_in[4];
  const float* bl0      = (const float*)d_in[5];
  const float* Wr0      = (const float*)d_in[6];
  const float* Wl1      = (const float*)d_in[7];
  const float* bl1      = (const float*)d_in[8];
  const float* Wr1      = (const float*)d_in[9];
  const float* Wout     = (const float*)d_in[10];
  const float* bout     = (const float*)d_in[11];
  float* out = (float*)d_out;

  char* base = (char*)d_ws;
  size_t off = 0;
  auto alloc = [&](size_t bytes) -> void* {
    void* p = base + off;
    off += (bytes + 255) & ~(size_t)255;
    return p;
  };
  float*          x     = (float*)alloc((size_t)NN * HIDD * 4);
  float*          s     = (float*)alloc((size_t)NN * HIDD * 4);
  float*          cnt   = (float*)alloc((size_t)NN * 4);
  unsigned short* meanb = (unsigned short*)alloc((size_t)NN * HIDD * 2);
  unsigned short* xb    = (unsigned short*)alloc((size_t)NN * HIDD * 2);
  unsigned short* WlT0  = (unsigned short*)alloc((size_t)HIDD * HIDD * 2);
  unsigned short* WrT0  = (unsigned short*)alloc((size_t)HIDD * HIDD * 2);
  unsigned short* WlT1  = (unsigned short*)alloc((size_t)HIDD * HIDD * 2);
  unsigned short* WrT1  = (unsigned short*)alloc((size_t)HIDD * HIDD * 2);
  float*          pooled= (float*)alloc((size_t)NG * HIDD * 4);
  float*          gcnt  = (float*)alloc((size_t)NG * 4);

  const int TPB = 256;
  auto ceildiv = [](int a, int b) { return (a + b - 1) / b; };

  prep_weights<<<ceildiv(HIDD * HIDD, TPB), TPB, 0, stream>>>(
      Wl0, Wr0, Wl1, Wr1, WlT0, WrT0, WlT1, WrT1);
  gather_embed<<<ceildiv(NN * EMBD, TPB), TPB, 0, stream>>>(node_ids, emb, x);

  const unsigned short* WlT[2] = {WlT0, WlT1};
  const unsigned short* WrT[2] = {WrT0, WrT1};
  const float*          bl[2]  = {bl0, bl1};

  const int gemm_blocks = 98;                       // 392 waves, ~8 tiles each
  const int gemm_waves  = gemm_blocks * 4;

  for (int layer = 0; layer < 2; ++layer) {
    zero_f32<<<ceildiv(NN * HIDD, TPB), TPB, 0, stream>>>(s, NN * HIDD);
    zero_f32<<<ceildiv(NN, TPB), TPB, 0, stream>>>(cnt, NN);
    scatter_edges<<<ceildiv(NE * 16, TPB), TPB, 0, stream>>>(edge, edge + NE, x, s, cnt);
    mean_bf_kernel<<<ceildiv(NN * HIDD, TPB), TPB, 0, stream>>>(s, cnt, x, meanb, xb);
    gemm_sage<<<gemm_blocks, 128, 0, stream>>>(meanb, xb, WlT[layer], WrT[layer],
                                               bl[layer], x, gemm_waves);
  }

  zero_f32<<<ceildiv(NG * HIDD, TPB), TPB, 0, stream>>>(pooled, NG * HIDD);
  zero_f32<<<ceildiv(NG, TPB), TPB, 0, stream>>>(gcnt, NG);
  pool_scatter<<<ceildiv(NN * 16, TPB), TPB, 0, stream>>>(batch, x, pooled, gcnt);
  final_out<<<NG, 32, 0, stream>>>(pooled, gcnt, Wout, bout, out);
}
